// Attention_31275951849847
// MI455X (gfx1250) — matrix operand order repaired and checked by reading, not compiled, not run on hardware
//
#include <hip/hip_runtime.h>
#include <stdint.h>

// Problem constants (from reference)
#define B_    2
#define NSEQ  2048
#define C_    1152
#define H_    16
#define HD_   72
#define HDP   96                    // head dim padded to 3 * 32 for WMMA K-chunks
#define SCALE_ 0.1178511301977579f  // 72^-0.5
#define NEG_  (-100000000.0f)

typedef __bf16 bf16_t;
typedef __attribute__((ext_vector_type(16))) __bf16 v16bf;
typedef __attribute__((ext_vector_type(8)))  __bf16 v8bf;
typedef __attribute__((ext_vector_type(8)))  float  v8f;

// ---- WMMA helpers ---------------------------------------------------------

static __device__ inline v8f wmma_bf16(v16bf a, v16bf b, v8f c) {
  // D = A(16x32 bf16) * B(32x16 bf16) + C(16x16 f32)
  return __builtin_amdgcn_wmma_f32_16x16x32_bf16(false, a, false, b,
                                                 (short)0, c, false, false);
}

// A-matrix 16x32 bf16 layout: lane l holds row m=l&15;
// elements 0..15 = K (l>>4)*16 .. +15 (contiguous in memory)
static __device__ inline v16bf load_a16(const bf16_t* p_lo, const bf16_t* p_hi) {
  (void)p_hi;
  const bf16_t* p = p_lo + ((threadIdx.x >> 4) & 1) * 8;
  v8bf lo = *reinterpret_cast<const v8bf*>(p);
  v8bf hi = *reinterpret_cast<const v8bf*>(p + 8);
  return __builtin_shufflevector(lo, hi, 0,1,2,3,4,5,6,7,8,9,10,11,12,13,14,15);
}

// Row reductions across the 16-lane half holding one C/D row
static __device__ inline float half_max16(float v) {
  v = fmaxf(v, __shfl_xor(v, 1, 32));
  v = fmaxf(v, __shfl_xor(v, 2, 32));
  v = fmaxf(v, __shfl_xor(v, 4, 32));
  v = fmaxf(v, __shfl_xor(v, 8, 32));
  return v;
}
static __device__ inline float half_sum16(float v) {
  v += __shfl_xor(v, 1, 32);
  v += __shfl_xor(v, 2, 32);
  v += __shfl_xor(v, 4, 32);
  v += __shfl_xor(v, 8, 32);
  return v;
}

// ---- Utility kernels ------------------------------------------------------

__global__ __launch_bounds__(256) void cvt_kernel(const float* __restrict__ src,
                                                  bf16_t* __restrict__ dst, int n) {
  int i = blockIdx.x * blockDim.x + threadIdx.x;
  if (i < n) dst[i] = (bf16_t)src[i];
}

__global__ __launch_bounds__(256) void zero_kernel(uint32_t* __restrict__ p, int nwords) {
  int i = blockIdx.x * blockDim.x + threadIdx.x;
  if (i < nwords) p[i] = 0u;
}

// ---- Kernel 1: QKV GEMM  qkv = x @ w_qkv.T --------------------------------
// One wave per 32x64 output block (2 A-frags x 4 B-frags, 8 accumulators).
__global__ __launch_bounds__(128) void qkv_kernel(const bf16_t* __restrict__ xb,
                                                  const bf16_t* __restrict__ wqb,
                                                  bf16_t* __restrict__ Qb,
                                                  bf16_t* __restrict__ Kb,
                                                  bf16_t* __restrict__ Vt) {
  const int lane = threadIdx.x & 31;
  const int wid  = (blockIdx.x * blockDim.x + threadIdx.x) >> 5;
  const int NT   = (3 * C_) / 64;          // 54 column blocks of 64
  const int nt   = wid % NT;
  const int mt   = wid / NT;               // 0..127 row blocks of 32 over B*N
  const int col  = lane & 15;
  const int half = lane >> 4;

  const v8f vzero = {0.f,0.f,0.f,0.f,0.f,0.f,0.f,0.f};
  v8f acc[2][4];
  for (int mi = 0; mi < 2; ++mi)
    for (int ni = 0; ni < 4; ++ni) acc[mi][ni] = vzero;

  const bf16_t* arow0 = xb + ((size_t)(mt * 32 + col)) * C_;       // A row m = lane&15
  const bf16_t* arow1 = arow0 + (size_t)16 * C_;
  const bf16_t* brow[4];
  for (int ni = 0; ni < 4; ++ni)
    brow[ni] = wqb + ((size_t)(nt * 64 + ni * 16 + col)) * C_;     // contraction contiguous

  for (int kc = 0; kc < C_ / 32; ++kc) {
    v16bf a0 = load_a16(arow0 + kc * 32 + half * 8, arow0 + kc * 32 + 16 + half * 8);
    v16bf a1 = load_a16(arow1 + kc * 32 + half * 8, arow1 + kc * 32 + 16 + half * 8);
    for (int ni = 0; ni < 4; ++ni) {
      v16bf bb = *reinterpret_cast<const v16bf*>(brow[ni] + kc * 32 + half * 16);
      acc[0][ni] = wmma_bf16(a0, bb, acc[0][ni]);
      acc[1][ni] = wmma_bf16(a1, bb, acc[1][ni]);
    }
  }

  // scatter D tiles: column n -> (s, h, hd); row m -> (b, seq)
  for (int ni = 0; ni < 4; ++ni) {
    const int n   = nt * 64 + ni * 16 + col;
    const int s   = n / C_;
    const int rem = n - s * C_;
    const int h   = rem / HD_;
    const int hd  = rem - h * HD_;
    for (int mi = 0; mi < 2; ++mi) {
      for (int j = 0; j < 8; ++j) {
        const int m  = mt * 32 + mi * 16 + half * 8 + j;   // global row over B*N
        const int b  = m / NSEQ;
        const int nq = m - b * NSEQ;
        const bf16_t val = (bf16_t)acc[mi][ni][j];
        if (s == 2) {
          Vt[(((size_t)(b * H_ + h)) * HDP + hd) * NSEQ + nq] = val;
        } else {
          const size_t idx = (((size_t)(b * H_ + h)) * NSEQ + nq) * HDP + hd;
          if (s == 0) Qb[idx] = val; else Kb[idx] = val;
        }
      }
    }
  }
}

// ---- Kernel 2: flash attention per (b,h,q-tile) wave, 64 keys/iter --------
__global__ __launch_bounds__(128) void attn_kernel(const bf16_t* __restrict__ Qb,
                                                   const bf16_t* __restrict__ Kb,
                                                   const bf16_t* __restrict__ Vt,
                                                   const int* __restrict__ mask,
                                                   bf16_t* __restrict__ ao) {
  __shared__ bf16_t lds[4][16 * 64];         // per-wave P tile (16 q-rows x 64 keys)
  const int lane = threadIdx.x & 31;
  const int wave = threadIdx.x >> 5;
  const int unit = blockIdx.x * 4 + wave;    // grid sized exactly
  const int QT   = NSEQ / 16;
  const int qt   = unit % QT;
  const int bh   = unit / QT;
  const int h    = bh % H_;
  const int b    = bh / H_;
  const int col  = lane & 15;
  const int half = lane >> 4;

  // Q tile as A-regs for 3 K=32 chunks over padded head dim
  const bf16_t* qrow = Qb + ((size_t)bh * NSEQ + qt * 16 + col) * HDP;
  v16bf aq[3];
  for (int c = 0; c < 3; ++c)
    aq[c] = load_a16(qrow + c * 32 + half * 8, qrow + c * 32 + 16 + half * 8);

  // per-row (C/D layout row = half*8 + j) query mask
  int mrow[8];
  for (int j = 0; j < 8; ++j) mrow[j] = mask[b * NSEQ + qt * 16 + half * 8 + j];

  float rmax[8], rsum[8], alph[8];
  for (int j = 0; j < 8; ++j) { rmax[j] = -3.0e38f; rsum[j] = 0.f; }
  const v8f vzero = {0.f,0.f,0.f,0.f,0.f,0.f,0.f,0.f};
  v8f acc[5];
  for (int ct = 0; ct < 5; ++ct) acc[ct] = vzero;   // hd 0..79 (only <72 stored)

  const bf16_t* Kbase = Kb + (size_t)bh * NSEQ * HDP;
  const bf16_t* Vbase = Vt + (size_t)bh * HDP * NSEQ;
  const int*    cmask = mask + b * NSEQ;
  bf16_t*       pl    = lds[wave];

  for (int kb = 0; kb < NSEQ / 64; ++kb) {      // 64 keys per iteration
    // ---- S = Q K^T for four 16-key subtiles
    v8f s[4];
    for (int t = 0; t < 4; ++t) s[t] = vzero;
    for (int c = 0; c < 3; ++c) {
      for (int t = 0; t < 4; ++t) {
        const bf16_t* kp = Kbase + ((size_t)(kb * 64 + t * 16) + col) * HDP + half * 16;
        v16bf bk = *reinterpret_cast<const v16bf*>(kp + c * 32);
        s[t] = wmma_bf16(aq[c], bk, s[t]);
      }
    }

    // ---- mask bias + online softmax over the 64-key block; write P to LDS
    int mk[4];
    for (int t = 0; t < 4; ++t) mk[t] = cmask[kb * 64 + t * 16 + col];
    for (int j = 0; j < 8; ++j) {
      float v[4];
      for (int t = 0; t < 4; ++t)
        v[t] = s[t][j] * SCALE_ + ((mk[t] & mrow[j]) ? 0.f : NEG_);
      float tm = fmaxf(fmaxf(v[0], v[1]), fmaxf(v[2], v[3]));
      tm = half_max16(tm);
      const float nm = fmaxf(rmax[j], tm);
      const float al = __expf(rmax[j] - nm);
      float p[4], psum = 0.f;
      for (int t = 0; t < 4; ++t) { p[t] = __expf(v[t] - nm); psum += p[t]; }
      const float ts = half_sum16(psum);
      rsum[j] = rsum[j] * al + ts;
      rmax[j] = nm;
      alph[j] = al;
      const int m = half * 8 + j;
      for (int t = 0; t < 4; ++t)
        pl[m * 64 + t * 16 + col] = (bf16_t)p[t];
    }
    for (int ct = 0; ct < 5; ++ct)
      for (int j = 0; j < 8; ++j) acc[ct][j] *= alph[j];

    __syncthreads();
    // ---- P (row-major 16x64 in LDS) -> A-frags for two K=32 chunks
    const bf16_t* pr = pl + (size_t)col * 64;    // A row m = lane&15
    v16bf ap[2];
    for (int c2 = 0; c2 < 2; ++c2)
      ap[c2] = load_a16(pr + c2 * 32 + half * 8, pr + c2 * 32 + 16 + half * 8);
    __syncthreads();

    // ---- acc += P @ V  (Vt column-contiguous per lane)
    for (int ct = 0; ct < 5; ++ct) {
      const int hd = ct * 16 + col;
      const bf16_t* vp = Vbase + (size_t)hd * NSEQ + kb * 64 + half * 16;
      for (int c2 = 0; c2 < 2; ++c2) {
        v16bf bv = *reinterpret_cast<const v16bf*>(vp + c2 * 32);
        acc[ct] = wmma_bf16(ap[c2], bv, acc[ct]);
      }
    }
  }

  // ---- normalize and store (bf16, layout ao[b][n][h*72+hd])
  for (int ct = 0; ct < 5; ++ct) {
    const int hd = ct * 16 + col;
    if (hd < HD_) {
      for (int j = 0; j < 8; ++j) {
        const int nq = qt * 16 + half * 8 + j;
        const float o = (rsum[j] > 0.f) ? acc[ct][j] / rsum[j] : 0.f;
        ao[((size_t)b * NSEQ + nq) * C_ + h * HD_ + hd] = (bf16_t)o;
      }
    }
  }
}

// ---- Kernel 3: out = ao @ w_proj.T + b_proj (f32 out), 32x64 blocking -----
__global__ __launch_bounds__(128) void proj_kernel(const bf16_t* __restrict__ ao,
                                                   const bf16_t* __restrict__ wpb,
                                                   const float* __restrict__ bp,
                                                   float* __restrict__ out) {
  const int lane = threadIdx.x & 31;
  const int wid  = (blockIdx.x * blockDim.x + threadIdx.x) >> 5;
  const int NT   = C_ / 64;                  // 18 column blocks
  const int nt   = wid % NT;
  const int mt   = wid / NT;                 // 0..127
  const int col  = lane & 15;
  const int half = lane >> 4;

  const v8f vzero = {0.f,0.f,0.f,0.f,0.f,0.f,0.f,0.f};
  v8f acc[2][4];
  for (int mi = 0; mi < 2; ++mi)
    for (int ni = 0; ni < 4; ++ni) acc[mi][ni] = vzero;

  const bf16_t* arow0 = ao + ((size_t)(mt * 32 + col)) * C_;
  const bf16_t* arow1 = arow0 + (size_t)16 * C_;
  const bf16_t* brow[4];
  for (int ni = 0; ni < 4; ++ni)
    brow[ni] = wpb + ((size_t)(nt * 64 + ni * 16 + col)) * C_;

  for (int kc = 0; kc < C_ / 32; ++kc) {
    v16bf a0 = load_a16(arow0 + kc * 32 + half * 8, arow0 + kc * 32 + 16 + half * 8);
    v16bf a1 = load_a16(arow1 + kc * 32 + half * 8, arow1 + kc * 32 + 16 + half * 8);
    for (int ni = 0; ni < 4; ++ni) {
      v16bf bb = *reinterpret_cast<const v16bf*>(brow[ni] + kc * 32 + half * 16);
      acc[0][ni] = wmma_bf16(a0, bb, acc[0][ni]);
      acc[1][ni] = wmma_bf16(a1, bb, acc[1][ni]);
    }
  }

  for (int ni = 0; ni < 4; ++ni) {
    const int n  = nt * 64 + ni * 16 + col;
    const float bn = bp[n];
    for (int mi = 0; mi < 2; ++mi) {
      for (int j = 0; j < 8; ++j) {
        const int m = mt * 32 + mi * 16 + half * 8 + j;
        out[(size_t)m * C_ + n] = acc[mi][ni][j] + bn;
      }
    }
  }
}

// ---- Launcher -------------------------------------------------------------

extern "C" void kernel_launch(void* const* d_in, const int* in_sizes, int n_in,
                              void* d_out, int out_size, void* d_ws, size_t ws_size,
                              hipStream_t stream) {
  (void)in_sizes; (void)n_in; (void)out_size; (void)ws_size;
  const float* x   = (const float*)d_in[0];
  const int*   msk = (const int*)d_in[1];
  const float* wq  = (const float*)d_in[2];
  const float* wp  = (const float*)d_in[3];
  const float* bp  = (const float*)d_in[4];
  float*       out = (float*)d_out;

  char* ws = (char*)d_ws;
  size_t off = 0;
  bf16_t* xb  = (bf16_t*)(ws + off); off += (size_t)B_ * NSEQ * C_ * 2;       // 9.4 MB (reused as ao)
  bf16_t* wqb = (bf16_t*)(ws + off); off += (size_t)3 * C_ * C_ * 2;          // 8.0 MB
  bf16_t* wpb = (bf16_t*)(ws + off); off += (size_t)C_ * C_ * 2;              // 2.7 MB
  bf16_t* Qb  = (bf16_t*)(ws + off); off += (size_t)B_ * H_ * NSEQ * HDP * 2; // 12.6 MB
  bf16_t* Kb  = (bf16_t*)(ws + off); off += (size_t)B_ * H_ * NSEQ * HDP * 2; // 12.6 MB
  bf16_t* Vt  = (bf16_t*)(ws + off); off += (size_t)B_ * H_ * HDP * NSEQ * 2; // 12.6 MB
  bf16_t* ao  = xb;  // xb dead after qkv_kernel; reuse for attention output

  const int nx  = B_ * NSEQ * C_;
  const int nwq = 3 * C_ * C_;
  const int nwp = C_ * C_;
  cvt_kernel<<<(nx  + 255) / 256, 256, 0, stream>>>(x,  xb,  nx);
  cvt_kernel<<<(nwq + 255) / 256, 256, 0, stream>>>(wq, wqb, nwq);
  cvt_kernel<<<(nwp + 255) / 256, 256, 0, stream>>>(wp, wpb, nwp);

  // zero Q/K/Vt (contiguous region) so padded head-dim lanes are exact zeros
  const int zwords = (int)(((size_t)3 * B_ * H_ * NSEQ * HDP * 2) / 4);
  zero_kernel<<<(zwords + 255) / 256, 256, 0, stream>>>((uint32_t*)Qb, zwords);

  // QKV GEMM: 128 x 54 blocks of 32x64, 4 waves/block
  qkv_kernel<<<(128 * 54) / 4, 128, 0, stream>>>(xb, wqb, Qb, Kb, Vt);

  // Attention: B*H*(N/16) = 4096 waves, 4 waves/block
  attn_kernel<<<(B_ * H_ * (NSEQ / 16)) / 4, 128, 0, stream>>>(Qb, Kb, Vt, msk, ao);

  // Projection: 128 x 18 blocks of 32x64, 4 waves/block
  proj_kernel<<<(128 * 18) / 4, 128, 0, stream>>>(ao, wpb, bp, out);
}
